// CrossAttention_12584254178060
// MI455X (gfx1250) — matrix operand order repaired and checked by reading, not compiled, not run on hardware
//
#include <hip/hip_runtime.h>
#include <hip/hip_bf16.h>
#include <math.h>
#include <stdint.h>

// Problem constants (from reference)
#define B_   4
#define DIM_ 256
#define NQ_  2048
#define MK_  2048
#define H_   8
#define DH_  64
#define DI_  512   // HEADS * DIM_HEAD

typedef __attribute__((ext_vector_type(16))) _Float16 v16h;
typedef __attribute__((ext_vector_type(8)))  _Float16 h8;    // 16-byte chunk
typedef __attribute__((ext_vector_type(8)))  float    v8f;
typedef __attribute__((ext_vector_type(4)))  unsigned int u32x4;
typedef __attribute__((ext_vector_type(8)))  int      i32x8;
typedef __attribute__((ext_vector_type(4)))  int      i32x4;

#if defined(__has_builtin)
#if __has_builtin(__builtin_amdgcn_tensor_load_to_lds) && \
    __has_builtin(__builtin_amdgcn_s_wait_tensorcnt)
#define USE_TDM 1
#endif
#endif
#ifndef USE_TDM
#define USE_TDM 0
#endif

static __device__ __forceinline__ v8f zero8() {
  v8f z = {0.f, 0.f, 0.f, 0.f, 0.f, 0.f, 0.f, 0.f};
  return z;
}

// ---- WMMA fragment loaders (layouts per CDNA5 ISA §7.12.2) ----
// A (16x32 f16, M x K), src row-major, leading dim ld (K contiguous).
static __device__ __forceinline__ v16h load_a16(const _Float16* src, int ld,
                                                int row0, int k0) {
  const int lane = threadIdx.x & 31;
  const int m  = row0 + (lane & 15);
  const int kb = (lane >> 4) * 8;
  const _Float16* p = src + (size_t)m * ld + k0 + kb;
  h8 lo = *(const h8*)(p);        // K = k0+kb .. +7
  h8 hi = *(const h8*)(p + 16);   // K = k0+16+kb .. +7
  v16h a;
#pragma unroll
  for (int j = 0; j < 8; ++j) { a[j] = lo[j]; a[j + 8] = hi[j]; }
  return a;
}

// B (32x16 f16, K x N) supplied TRANSPOSED: srcT is N-major (row n holds K
// contiguous). lane L: col = n0 + (L&15); K = k0 + (L>>4)*8 + j, +16 for j>=8.
static __device__ __forceinline__ v16h load_bt16(const _Float16* srcT, int ldT,
                                                 int n0, int k0) {
  const int lane = threadIdx.x & 31;
  const int n  = n0 + (lane & 15);
  const int kb = (lane >> 4) * 8;
  const _Float16* p = srcT + (size_t)n * ldT + k0 + kb;
  h8 lo = *(const h8*)(p);
  h8 hi = *(const h8*)(p + 16);
  v16h b;
#pragma unroll
  for (int j = 0; j < 8; ++j) { b[j] = lo[j]; b[j + 8] = hi[j]; }
  return b;
}

static __device__ __forceinline__ v8f wmma16(v16h a, v16h b, v8f c) {
  return __builtin_amdgcn_wmma_f32_16x16x32_f16(false, a, false, b,
                                                (short)0, c, false, false);
}

#if USE_TDM
// Async 2D tile DMA global->LDS via Tensor Data Mover.
// Descriptor layout per CDNA5 ISA (D# groups).
// All fields wave-uniform (readfirstlane). data_size = 2 bytes.
static __device__ __forceinline__ void tdm_load_2d(unsigned lds_off,
                                                   unsigned long long gaddr,
                                                   unsigned td0, unsigned td1,
                                                   unsigned long long stride0,
                                                   unsigned tile0, unsigned tile1) {
  u32x4 g0;
  g0[0] = (unsigned)__builtin_amdgcn_readfirstlane(1);  // count=1, user mode
  g0[1] = (unsigned)__builtin_amdgcn_readfirstlane((int)lds_off);
  g0[2] = (unsigned)__builtin_amdgcn_readfirstlane((int)(unsigned)(gaddr & 0xffffffffull));
  g0[3] = (unsigned)__builtin_amdgcn_readfirstlane(
      (int)(((unsigned)(gaddr >> 32) & 0x01ffffffu) | (2u << 30)));  // type=2
  i32x8 g1;
  g1[0] = __builtin_amdgcn_readfirstlane((int)(1u << 16));  // data_size=1 -> 2B
  g1[1] = __builtin_amdgcn_readfirstlane((int)((td0 & 0xffffu) << 16));
  g1[2] = __builtin_amdgcn_readfirstlane(
      (int)(((td0 >> 16) & 0xffffu) | ((td1 & 0xffffu) << 16)));
  g1[3] = __builtin_amdgcn_readfirstlane(
      (int)(((td1 >> 16) & 0xffffu) | ((tile0 & 0xffffu) << 16)));
  g1[4] = __builtin_amdgcn_readfirstlane((int)(tile1 & 0xffffu));  // tile_dim2=0
  g1[5] = __builtin_amdgcn_readfirstlane((int)(unsigned)(stride0 & 0xffffffffull));
  g1[6] = __builtin_amdgcn_readfirstlane((int)(unsigned)((stride0 >> 32) & 0xffffu));
  g1[7] = __builtin_amdgcn_readfirstlane(0);
  i32x4 gz = {0, 0, 0, 0};
#if defined(__clang_major__) && (__clang_major__ >= 23)
  i32x8 gz8 = {0, 0, 0, 0, 0, 0, 0, 0};
  __builtin_amdgcn_tensor_load_to_lds(g0, g1, gz, gz, gz8, 0);
#else
  __builtin_amdgcn_tensor_load_to_lds(g0, g1, gz, gz, 0);
#endif
}
#endif  // USE_TDM

// ---------------- prep kernels ----------------
__global__ void cast_f16_kernel(const float* __restrict__ src,
                                _Float16* __restrict__ dst, int n) {
  for (int i = blockIdx.x * blockDim.x + threadIdx.x; i < n;
       i += gridDim.x * blockDim.x)
    dst[i] = (_Float16)src[i];
}

// x: [B, C, Ncols] f32 -> y: [B, Ncols, C] f16 normalized over C
__global__ void rmsnorm_cast_kernel(const float* __restrict__ x,
                                    const float* __restrict__ gamma,
                                    _Float16* __restrict__ y,
                                    int C, int Ncols) {
  int idx = blockIdx.x * blockDim.x + threadIdx.x;   // over B*Ncols
  int b = idx / Ncols, n = idx % Ncols;
  const float* xb = x + (size_t)b * C * Ncols + n;
  float ss = 0.f;
  for (int c = 0; c < C; ++c) { float v = xb[(size_t)c * Ncols]; ss += v * v; }
  float inv = 16.0f / fmaxf(sqrtf(ss), 1e-12f);      // sqrt(256)=16
  _Float16* yb = y + ((size_t)b * Ncols + n) * C;
  for (int c = 0; c < C; ++c)
    yb[c] = (_Float16)(xb[(size_t)c * Ncols] * inv * gamma[c]);
}

// Y = X * W^T via WMMA. Each wave computes a 16x64 strip: one A fragment is
// reused by 4 WMMAs (compute-bound kernel -> maximize wmma per byte loaded).
__global__ void gemm_nt_kernel(const _Float16* __restrict__ X,
                               const _Float16* __restrict__ WT,
                               _Float16* __restrict__ Y,
                               int Nrows, int C, int O) {
  const int wave = threadIdx.x >> 5;
  const int lane = threadIdx.x & 31;
  const int nt = blockIdx.x;
  const int og = (blockIdx.y * 4 + wave) * 4;   // 4 o-tiles per wave
  const int b  = blockIdx.z;
  if (og * 16 >= O) return;
  const _Float16* A = X + (size_t)b * Nrows * C;
  v8f acc[4] = {zero8(), zero8(), zero8(), zero8()};
  for (int k0 = 0; k0 < C; k0 += 32) {
    v16h a = load_a16(A, C, nt * 16, k0);
#pragma unroll
    for (int t = 0; t < 4; ++t)
      acc[t] = wmma16(a, load_bt16(WT, C, (og + t) * 16, k0), acc[t]);
  }
  const int rb = nt * 16 + (lane >> 4) * 8;
#pragma unroll
  for (int t = 0; t < 4; ++t) {
    const int col = (og + t) * 16 + (lane & 15);
#pragma unroll
    for (int r = 0; r < 8; ++r)
      Y[((size_t)b * Nrows + rb + r) * O + col] = (_Float16)acc[t][r];
  }
}

// sq[(b*H+h)*Nrows + n] = sum_d Y[b][n][obase + h*64 + d]^2
__global__ void rowsq_kernel(const _Float16* __restrict__ Y,
                             float* __restrict__ sq,
                             int Nrows, int O, int obase, int total) {
  int idx = blockIdx.x * blockDim.x + threadIdx.x;
  if (idx >= total) return;
  int n = idx % Nrows, rest = idx / Nrows;
  int h = rest % H_, b = rest / H_;
  const _Float16* p = Y + ((size_t)b * Nrows + n) * O + obase + h * DH_;
  float s = 0.f;
#pragma unroll 8
  for (int d = 0; d < DH_; ++d) { float v = (float)p[d]; s += v * v; }
  sq[idx] = s;
}

// vT[b][h][d][m] = KV[b][m][512 + h*64 + d]
__global__ void vtrans_kernel(const _Float16* __restrict__ KV,
                              _Float16* __restrict__ vT, int total) {
  int idx = blockIdx.x * blockDim.x + threadIdx.x;
  if (idx >= total) return;
  int m = idx % MK_, rest = idx / MK_;
  int d = rest % DH_; rest /= DH_;
  int h = rest % H_, b = rest / H_;
  vT[idx] = KV[((size_t)b * MK_ + m) * (2 * DI_) + DI_ + h * DH_ + d];
}

// ---------------- flash L2-distance attention ----------------
// 1 wave per (b, h, 16-row q tile). K/V chunks staged into per-wave LDS
// double buffers by the Tensor Data Mover, synced with s_wait_tensorcnt.
#define AW 4
__global__ void attn_kernel(const _Float16* __restrict__ Q,
                            const _Float16* __restrict__ KV,
                            const _Float16* __restrict__ vT,
                            const float* __restrict__ q2,
                            const float* __restrict__ k2,
                            const unsigned char* __restrict__ mask,
                            _Float16* __restrict__ AO) {
  __shared__ __align__(16) _Float16 lds_P[AW][16 * 32];
#if USE_TDM
  __shared__ __align__(16) _Float16 lds_K[AW][2][32 * 64];  // rows m, d contig
  __shared__ __align__(16) _Float16 lds_V[AW][2][64 * 32];  // rows d, m contig
#endif
  __shared__ float lds_q2[AW][16];
  __shared__ float lds_m[AW][16];
  __shared__ float lds_l[AW][16];

  const int wave = threadIdx.x >> 5;
  const int lane = threadIdx.x & 31;
  const int cl   = lane & 15;          // column within 16-wide tile
  const int rr0  = (lane >> 4) * 8;    // row base within tile for C layout
  const int ntiles = NQ_ / 16;
  int tile = blockIdx.x * AW + wave;
  if (tile >= B_ * H_ * ntiles) return;
  const int nt = tile % ntiles;
  const int h  = (tile / ntiles) % H_;
  const int b  = tile / (ntiles * H_);
  const int n0 = nt * 16;

  const _Float16* Qp  = Q  + (size_t)b * NQ_ * DI_ + h * DH_;        // ld=512
  const _Float16* Kp  = KV + (size_t)b * MK_ * (2 * DI_) + h * DH_;  // ld=1024
  const _Float16* Vp  = vT + ((size_t)(b * H_ + h) * DH_) * MK_;     // ld=2048
  const float* k2p = k2 + (size_t)(b * H_ + h) * MK_;
  const unsigned char* mp = mask + (size_t)b * MK_;

  const v16h aq0 = load_a16(Qp, DI_, n0, 0);
  const v16h aq1 = load_a16(Qp, DI_, n0, 32);
  if (lane < 16) {
    lds_q2[wave][lane] = q2[(size_t)(b * H_ + h) * NQ_ + n0 + lane];
    lds_m[wave][lane] = -3.0e38f;
    lds_l[wave][lane] = 0.0f;
  }
  // single-wave LDS usage: DS ops in program order within a wave, no barrier.

#if USE_TDM
  const unsigned ldsK_off =
      (unsigned)(unsigned long long)(uintptr_t)(void*)&lds_K[wave][0][0];
  const unsigned ldsV_off =
      (unsigned)(unsigned long long)(uintptr_t)(void*)&lds_V[wave][0][0];
  auto issue_chunk = [&](int mc, int bufn) {
    unsigned long long gK = (unsigned long long)(uintptr_t)(Kp + (size_t)mc * (2 * DI_));
    unsigned long long gV = (unsigned long long)(uintptr_t)(Vp + mc);
    // K: tile 64(d,contig) x 32(m rows), row stride 1024 elements
    tdm_load_2d(ldsK_off + (unsigned)bufn * 4096u, gK, 64u, 32u, 1024ull, 64u, 32u);
    // V: tile 32(m,contig) x 64(d rows), row stride 2048 elements
    tdm_load_2d(ldsV_off + (unsigned)bufn * 4096u, gV, 32u, 64u, 2048ull, 32u, 64u);
  };
  issue_chunk(0, 0);
#endif

  v8f oacc[4];
#pragma unroll
  for (int t = 0; t < 4; ++t) oacc[t] = zero8();
  const float ascale = 0.125f;  // DIM_HEAD^-0.5

  for (int m0 = 0; m0 < MK_; m0 += 32) {
#if USE_TDM
    const int buf = (m0 >> 5) & 1;
    if (m0 + 32 < MK_) {
      issue_chunk(m0 + 32, buf ^ 1);
      __builtin_amdgcn_s_wait_tensorcnt((short)2);  // chunk m0 complete
    } else {
      __builtin_amdgcn_s_wait_tensorcnt((short)0);
    }
    asm volatile("" ::: "memory");  // keep ds reads below the tensorcnt wait
    const _Float16* kb_ = &lds_K[wave][buf][0];
    const _Float16* vb_ = &lds_V[wave][buf][0];
    v8f sA = zero8(), sB = zero8();
    sA = wmma16(aq0, load_bt16(kb_, 64, 0, 0), sA);
    sA = wmma16(aq1, load_bt16(kb_, 64, 0, 32), sA);
    sB = wmma16(aq0, load_bt16(kb_, 64, 16, 0), sB);
    sB = wmma16(aq1, load_bt16(kb_, 64, 16, 32), sB);
#else
    if (m0 + 32 < MK_) {  // global_prefetch_b8 of next chunk
      __builtin_prefetch(Kp + (size_t)(m0 + 32 + cl) * (2 * DI_), 0, 1);
      __builtin_prefetch(Vp + (size_t)cl * MK_ + m0 + 32, 0, 1);
    }
    v8f sA = zero8(), sB = zero8();
    sA = wmma16(aq0, load_bt16(Kp, 2 * DI_, m0, 0), sA);
    sA = wmma16(aq1, load_bt16(Kp, 2 * DI_, m0, 32), sA);
    sB = wmma16(aq0, load_bt16(Kp, 2 * DI_, m0 + 16, 0), sB);
    sB = wmma16(aq1, load_bt16(Kp, 2 * DI_, m0 + 16, 32), sB);
#endif

    const float k2A = k2p[m0 + cl],     k2B = k2p[m0 + 16 + cl];
    const bool  mkA = mp[m0 + cl] != 0, mkB = mp[m0 + 16 + cl] != 0;
    float sc[8];
#pragma unroll
    for (int r = 0; r < 8; ++r) {
      float q2v = lds_q2[wave][rr0 + r];
      float dA = fmaxf(q2v + k2A - 2.0f * sA[r], 0.0f);
      float dB = fmaxf(q2v + k2B - 2.0f * sB[r], 0.0f);
      float simA = mkA ? -sqrtf(fmaxf(dA, 1e-12f)) * ascale : -3.0e38f;
      float simB = mkB ? -sqrtf(fmaxf(dB, 1e-12f)) * ascale : -3.0e38f;
      float rowm = fmaxf(simA, simB);
#pragma unroll
      for (int off = 8; off > 0; off >>= 1)
        rowm = fmaxf(rowm, __shfl_xor(rowm, off, 16));
      float oldm = lds_m[wave][rr0 + r];
      float newm = fmaxf(oldm, rowm);
      float pA = __expf(simA - newm);
      float pB = __expf(simB - newm);
      float rs = pA + pB;
#pragma unroll
      for (int off = 8; off > 0; off >>= 1) rs += __shfl_xor(rs, off, 16);
      sc[r] = __expf(oldm - newm);
      if (cl == 0) {
        lds_m[wave][rr0 + r] = newm;
        lds_l[wave][rr0 + r] = lds_l[wave][rr0 + r] * sc[r] + rs;
      }
      // stage P (C layout -> row-major f16 in LDS for A-fragment reload)
      lds_P[wave][(rr0 + r) * 32 + cl]      = (_Float16)pA;
      lds_P[wave][(rr0 + r) * 32 + 16 + cl] = (_Float16)pB;
    }
#pragma unroll
    for (int t = 0; t < 4; ++t)
#pragma unroll
      for (int r = 0; r < 8; ++r) oacc[t][r] *= sc[r];

    const v16h ap = load_a16(&lds_P[wave][0], 32, 0, 0);  // ds_load path
#pragma unroll
    for (int t = 0; t < 4; ++t) {
#if USE_TDM
      oacc[t] = wmma16(ap, load_bt16(vb_, 32, t * 16, 0), oacc[t]);
#else
      oacc[t] = wmma16(ap, load_bt16(Vp, MK_, t * 16, m0), oacc[t]);
#endif
    }
  }

  float invl[8];
#pragma unroll
  for (int r = 0; r < 8; ++r) invl[r] = 1.0f / lds_l[wave][rr0 + r];
#pragma unroll
  for (int t = 0; t < 4; ++t)
#pragma unroll
    for (int r = 0; r < 8; ++r) {
      size_t row = (size_t)n0 + rr0 + r;
      AO[((size_t)b * NQ_ + row) * DI_ + h * DH_ + t * 16 + cl] =
          (_Float16)(oacc[t][r] * invl[r]);
    }
}

// out[b][o2][n] = sum_o Wout[o2][o] * AO[b][n][o]   (f32 output)
__global__ void out_gemm_kernel(const _Float16* __restrict__ X,  // [B,N,512]
                                const _Float16* __restrict__ WoT, // [256,512]
                                float* __restrict__ out, int Nrows) {
  const int wave = threadIdx.x >> 5;
  const int lane = threadIdx.x & 31;
  const int nt = blockIdx.x;
  const int og = wave * 4;             // 4 o2-tiles per wave, 16 tiles total
  const int b  = blockIdx.z;
  const _Float16* A = X + (size_t)b * Nrows * DI_;
  v8f acc[4] = {zero8(), zero8(), zero8(), zero8()};
  for (int k0 = 0; k0 < DI_; k0 += 32) {
    v16h a = load_a16(A, DI_, nt * 16, k0);
#pragma unroll
    for (int t = 0; t < 4; ++t)
      acc[t] = wmma16(a, load_bt16(WoT, DI_, (og + t) * 16, k0), acc[t]);
  }
  const int rb = nt * 16 + (lane >> 4) * 8;
#pragma unroll
  for (int t = 0; t < 4; ++t) {
    const int col = (og + t) * 16 + (lane & 15);   // o2
#pragma unroll
    for (int r = 0; r < 8; ++r)
      out[((size_t)b * DIM_ + col) * Nrows + rb + r] = acc[t][r];
  }
}

extern "C" void kernel_launch(void* const* d_in, const int* in_sizes, int n_in,
                              void* d_out, int out_size, void* d_ws, size_t ws_size,
                              hipStream_t stream) {
  const float* fmap   = (const float*)d_in[0];
  const float* ctx    = (const float*)d_in[1];
  const unsigned char* mask = (const unsigned char*)d_in[2];  // jax bool = 1 byte
  const float* gamma  = (const float*)d_in[3];
  const float* gammac = (const float*)d_in[4];
  const float* Wq     = (const float*)d_in[5];
  const float* Wkv    = (const float*)d_in[6];
  const float* Wout   = (const float*)d_in[7];
  float* out = (float*)d_out;
  char* ws = (char*)d_ws;

  // workspace layout (all 256B-aligned offsets), ~49.5 MB total
  _Float16* WqH  = (_Float16*)(ws + 0);          //  512*256 f16
  _Float16* WkvH = (_Float16*)(ws + 262144);     // 1024*256 f16
  _Float16* WoH  = (_Float16*)(ws + 786432);     //  256*512 f16
  _Float16* Xn   = (_Float16*)(ws + 1048576);    // [B,N,256] f16
  _Float16* Cn   = (_Float16*)(ws + 5242880);    // [B,M,256] f16
  _Float16* Qb   = (_Float16*)(ws + 9437184);    // [B,N,512] f16
  _Float16* KVb  = (_Float16*)(ws + 17825792);   // [B,M,1024] f16
  _Float16* VT   = (_Float16*)(ws + 34603008);   // [B,H,64,M] f16
  float*    Q2   = (float*)(ws + 42991616);      // [B,H,N]
  float*    K2   = (float*)(ws + 43253760);      // [B,H,M]
  _Float16* AO   = (_Float16*)(ws + 43515904);   // [B,N,512] f16

  cast_f16_kernel<<<128, 256, 0, stream>>>(Wq,  WqH,  DI_ * DIM_);
  cast_f16_kernel<<<128, 256, 0, stream>>>(Wkv, WkvH, 2 * DI_ * DIM_);
  cast_f16_kernel<<<128, 256, 0, stream>>>(Wout, WoH, DIM_ * DI_);

  rmsnorm_cast_kernel<<<B_ * NQ_ / 256, 256, 0, stream>>>(fmap, gamma,  Xn, DIM_, NQ_);
  rmsnorm_cast_kernel<<<B_ * MK_ / 256, 256, 0, stream>>>(ctx,  gammac, Cn, DIM_, MK_);

  dim3 gq(NQ_ / 16, DI_ / 256, B_);        // (128, 2, 4)
  gemm_nt_kernel<<<gq, 128, 0, stream>>>(Xn, WqH, Qb, NQ_, DIM_, DI_);
  dim3 gkv(MK_ / 16, (2 * DI_) / 256, B_); // (128, 4, 4)
  gemm_nt_kernel<<<gkv, 128, 0, stream>>>(Cn, WkvH, KVb, MK_, DIM_, 2 * DI_);

  rowsq_kernel<<<(B_ * H_ * NQ_) / 256, 256, 0, stream>>>(Qb,  Q2, NQ_, DI_,     0, B_ * H_ * NQ_);
  rowsq_kernel<<<(B_ * H_ * MK_) / 256, 256, 0, stream>>>(KVb, K2, MK_, 2 * DI_, 0, B_ * H_ * MK_);
  vtrans_kernel<<<(B_ * H_ * DH_ * MK_) / 256, 256, 0, stream>>>(KVb, VT, B_ * H_ * DH_ * MK_);

  attn_kernel<<<(B_ * H_ * (NQ_ / 16)) / AW, 32 * AW, 0, stream>>>(
      Qb, KVb, VT, Q2, K2, mask, AO);

  dim3 go(NQ_ / 16, 1, B_);                // 16 o2-tiles = 4 waves x 4
  out_gemm_kernel<<<go, 128, 0, stream>>>(AO, WoH, out, NQ_);
}
